// GNNStack_53085795779206
// MI455X (gfx1250) — compile-verified, matrix operand order checked by reading, not run on hardware
//
#include <hip/hip_runtime.h>
#include <cstdint>
#include <cstddef>
#include <cmath>

#define PIf 3.14159265358979323846f

typedef __attribute__((ext_vector_type(16))) __bf16          v16bf;
typedef __attribute__((ext_vector_type(16))) unsigned short  v16u;
typedef __attribute__((ext_vector_type(8)))  float           v8f;
typedef __attribute__((ext_vector_type(4)))  unsigned int    v4u_;
typedef __attribute__((ext_vector_type(8)))  int             v8i_;
typedef __attribute__((ext_vector_type(4)))  int             v4i_;

#if defined(__has_builtin)
#if __has_builtin(__builtin_amdgcn_tensor_load_to_lds) && __has_builtin(__builtin_amdgcn_s_wait_tensorcnt)
#define USE_TDM 1
#endif
#endif
#ifndef USE_TDM
#define USE_TDM 0
#endif

__device__ __forceinline__ unsigned short f2bf(float f) {
    unsigned int u = __float_as_uint(f);
    unsigned int r = u + 0x7FFFu + ((u >> 16) & 1u);   // round-to-nearest-even
    return (unsigned short)(r >> 16);
}

__device__ __forceinline__ float sigm(float z) { return 1.f / (1.f + expf(-z)); }

// ---------------------------------------------------------------- utilities
__global__ void k_zero(float* p, long n) {
    long i = (long)blockIdx.x * blockDim.x + threadIdx.x;
    if (i < n) p[i] = 0.f;
}

// copy x[:,2:10] -> feat (N x 8)
__global__ void k_feat0(const float* __restrict__ x, float* __restrict__ feat, int N) {
    int i = blockIdx.x * blockDim.x + threadIdx.x;
    if (i < N * 8) {
        int n = i >> 3, c = i & 7;
        feat[i] = x[(size_t)n * 10 + 2 + c];
    }
}

// stats[0]=mean(eta), stats[1]=mean(phi), stats[2]=Sx=sum(pt*cos(phi))
__global__ void k_stats(const float* __restrict__ x, int N, float* __restrict__ stats) {
    __shared__ float se[256], sp[256], sx[256];
    float a = 0.f, b = 0.f, c = 0.f;
    for (int n = threadIdx.x; n < N; n += blockDim.x) {
        float eta = x[(size_t)n * 10 + 0];
        float phi = x[(size_t)n * 10 + 1];
        float pt  = x[(size_t)n * 10 + 2];
        a += eta; b += phi; c += pt * cosf(phi);
    }
    se[threadIdx.x] = a; sp[threadIdx.x] = b; sx[threadIdx.x] = c;
    __syncthreads();
    for (int s = 128; s > 0; s >>= 1) {
        if ((int)threadIdx.x < s) {
            se[threadIdx.x] += se[threadIdx.x + s];
            sp[threadIdx.x] += sp[threadIdx.x + s];
            sx[threadIdx.x] += sx[threadIdx.x + s];
        }
        __syncthreads();
    }
    if (threadIdx.x == 0) {
        stats[0] = se[0] / (float)N;
        stats[1] = sp[0] / (float)N;
        stats[2] = sx[0];
    }
}

// per-column mean of feat (N x C) -> xg[C]
__global__ void k_colmean(const float* __restrict__ feat, int N, int C, float* __restrict__ xg) {
    __shared__ float sh[256];
    int c = blockIdx.x;
    float s = 0.f;
    for (int n = threadIdx.x; n < N; n += blockDim.x) s += feat[(size_t)n * C + c];
    sh[threadIdx.x] = s;
    __syncthreads();
    for (int t = 128; t > 0; t >>= 1) {
        if ((int)threadIdx.x < t) sh[threadIdx.x] += sh[threadIdx.x + t];
        __syncthreads();
    }
    if (threadIdx.x == 0) xg[c] = sh[0] / (float)N;
}

// consts[0..2]=m2 geo weights, consts[3]=m2 constant part, consts[4]=m5 constant part
__global__ void k_prep(const float* __restrict__ w2, const float* __restrict__ b2,
                       const float* __restrict__ w5, const float* __restrict__ b5,
                       const float* __restrict__ xg, const float* __restrict__ stats,
                       int C, float logc, float* __restrict__ consts) {
    if (threadIdx.x != 0 || blockIdx.x != 0) return;
    int d = 3 * C + 4;
    float s = 0.f;
    for (int c = 0; c < C; ++c) s += w2[2 * C + c] * xg[c];
    consts[0] = w2[3 * C];
    consts[1] = w2[3 * C + 1];
    consts[2] = w2[3 * C + 2];
    consts[3] = s + w2[3 * C + 3] * logc + b2[0];
    float s5 = 0.f;
    for (int c = 0; c < C; ++c) {
        float xgu = (c < C - 2) ? xg[c + 2] : ((c == C - 2) ? stats[0] : stats[1]);
        s5 += w5[d + C + c] * xgu;
    }
    consts[4] = s5 + w5[d + 2 * C] * logc + b5[0];
}

// per-node partial m2 dots
__global__ void k_dotji(const float* __restrict__ feat, const float* __restrict__ w2,
                        int C, int N, float* __restrict__ dotJ, float* __restrict__ dotI) {
    int n = blockIdx.x * blockDim.x + threadIdx.x;
    if (n >= N) return;
    const float* f = feat + (size_t)n * C;
    float a = 0.f, b = 0.f;
    for (int c = 0; c < C; ++c) { float v = f[c]; a += w2[c] * v; b += w2[C + c] * v; }
    dotJ[n] = a; dotI[n] = b;
}

// heavy edge kernel: 2^tpeShift threads per edge; atomics into ssum (N x (C+4)) and cnt
__global__ void k_edge(const int* __restrict__ src, const int* __restrict__ dst, int E,
                       const float* __restrict__ x, const float* __restrict__ feat, int C,
                       const float* __restrict__ dotJ, const float* __restrict__ dotI,
                       const float* __restrict__ consts,
                       float* __restrict__ ssum, float* __restrict__ cnt, int tpeShift) {
    long tid = (long)blockIdx.x * blockDim.x + threadIdx.x;
    int e = (int)(tid >> tpeShift);
    if (e >= E) return;
    int sub = (int)(tid & ((1 << tpeShift) - 1));
    int s = src[e], t = dst[e];
    const float* fs = feat + (size_t)s * C;
    __builtin_prefetch(fs, 0, 0);                     // global_prefetch_b8
    float etas = x[(size_t)s * 10], phis = x[(size_t)s * 10 + 1];
    float etat = x[(size_t)t * 10], phit = x[(size_t)t * 10 + 1];
    float deta = etas - etat;
    float dphi = phis - phit;
    if (dphi > PIf) dphi -= ceilf((dphi - PIf) * (1.f / (2.f * PIf))) * (2.f * PIf);
    float dr = fabsf(deta) + fabsf(dphi);
    float gate = sigm(dotJ[s] + dotI[t] + consts[0] * deta + consts[1] * dphi
                      + consts[2] * dr + consts[3]);
    float* sr = ssum + (size_t)t * (C + 4);
    int tpe = 1 << tpeShift;
    for (int c = sub; c < C; c += tpe) unsafeAtomicAdd(&sr[c], gate * fs[c]);
    if (sub == 0) {
        unsafeAtomicAdd(&sr[C + 0], gate * deta);
        unsafeAtomicAdd(&sr[C + 1], gate * dphi);
        unsafeAtomicAdd(&sr[C + 2], gate * dr);
        unsafeAtomicAdd(&sr[C + 3], gate);
        unsafeAtomicAdd(&cnt[t], 1.f);
    }
}

// build aggr (bf16, N x Kp2) and x_u (bf16, N x Kp1), zero-padded, and the
// m5 gate dot-product partial zpart[n] (f32, exact values from registers)
__global__ void k_aggr(const float* __restrict__ ssum, const float* __restrict__ cnt,
                       const float* __restrict__ feat, const float* __restrict__ xg,
                       const float* __restrict__ x, const float* __restrict__ w5,
                       int C, int N, int Kp2, int Kp1, float logc,
                       unsigned short* __restrict__ aggr_bf,
                       unsigned short* __restrict__ xu_bf,
                       float* __restrict__ zpart) {
    int n = blockIdx.x * blockDim.x + threadIdx.x;
    if (n >= N) return;
    int d = 3 * C + 4;
    const float* sr = ssum + (size_t)n * (C + 4);
    const float* f  = feat + (size_t)n * C;
    float inv = 1.f / fmaxf(cnt[n], 1.f);
    float gs  = sr[C + 3] * inv;
    unsigned short* ar = aggr_bf + (size_t)n * Kp2;
    float z = 0.f;
    for (int c = 0; c < C; ++c) { float v = sr[c] * inv;  ar[c]         = f2bf(v); z += w5[c] * v; }
    for (int c = 0; c < C; ++c) { float v = f[c] * gs;    ar[C + c]     = f2bf(v); z += w5[C + c] * v; }
    for (int c = 0; c < C; ++c) { float v = xg[c] * gs;   ar[2 * C + c] = f2bf(v); z += w5[2 * C + c] * v; }
    {
        float v0 = sr[C] * inv, v1 = sr[C + 1] * inv, v2 = sr[C + 2] * inv, v3 = logc * gs;
        ar[3 * C]     = f2bf(v0); z += w5[3 * C]     * v0;
        ar[3 * C + 1] = f2bf(v1); z += w5[3 * C + 1] * v1;
        ar[3 * C + 2] = f2bf(v2); z += w5[3 * C + 2] * v2;
        ar[3 * C + 3] = f2bf(v3); z += w5[3 * C + 3] * v3;
    }
    for (int k = d; k < Kp2; ++k) ar[k] = 0;
    unsigned short* xr = xu_bf + (size_t)n * Kp1;
    for (int c = 0; c < C - 2; ++c) { float v = f[2 + c]; xr[c] = f2bf(v); z += w5[d + c] * v; }
    {
        float ve = x[(size_t)n * 10], vp = x[(size_t)n * 10 + 1];
        xr[C - 2] = f2bf(ve); z += w5[d + C - 2] * ve;
        xr[C - 1] = f2bf(vp); z += w5[d + C - 1] * vp;
    }
    for (int k = C; k < Kp1; ++k) xr[k] = 0;
    zpart[n] = z;
}

// pack W (64 x K row-major, f32) into bf16 WMMA B-fragment layout
__global__ void k_pack(const float* __restrict__ W, int K, int Ktiles,
                       unsigned short* __restrict__ Bp) {
    int idx = blockIdx.x * blockDim.x + threadIdx.x;
    int total = Ktiles * 2048;
    if (idx >= total) return;
    int slot = idx & 15, lane = (idx >> 4) & 31, j = (idx >> 9) & 3, kt = idx >> 11;
    int g = lane >> 4, n = lane & 15;
    int v = slot >> 1, p = slot & 1;
    int k = kt * 32 + ((v >= 4) ? 16 : 0) + g * 8 + (v & 3) * 2 + p;
    int col = j * 16 + n;
    float w = (k < K) ? W[(size_t)col * K + k] : 0.f;   // B[k][o] = W[o][k]
    Bp[idx] = f2bf(w);
}

// D(N x 64) = A_bf16(N x Kp) * Bpacked + bias, optional relu + optional bf16 copy.
// B staged in LDS once per workgroup via the Tensor Data Mover (fallback: coop copy).
template<int KT>
__global__ void k_gemm(const unsigned short* __restrict__ A, int lda,
                       const unsigned short* __restrict__ Bp,
                       const float* __restrict__ bias, int relu,
                       float* __restrict__ D, unsigned short* __restrict__ Dbf, int N) {
    __shared__ v16u Bs[KT * 4 * 32];
#if USE_TDM
    if ((threadIdx.x >> 5) == 0) {
        unsigned nd = (unsigned)(KT * 1024);              // payload in dwords
        unsigned long long ga = (unsigned long long)(uintptr_t)Bp;
        unsigned lds = (unsigned)(uintptr_t)(&Bs[0]);     // low 32 bits = LDS offset
        // D# group0: count=1, lds_addr, global_addr[56:0], type=2
        v4u_ g0 = { 1u, lds, (unsigned)(ga & 0xFFFFFFFFu),
                    (unsigned)((ga >> 32) & 0x1FFFFFFu) | (2u << 30) };
        // D# group1: data_size=4B; tensor_dim0=nd, tensor_dim1=1; tile 1 x nd;
        // tensor_dim0_stride=nd
        v8i_ g1 = { (int)(2u << 16),
                    (int)((nd & 0xFFFFu) << 16),
                    (int)((nd >> 16) | (1u << 16)),
                    (int)(nd << 16),
                    1,
                    (int)nd,
                    0, 0 };
        v4i_ gz4 = { 0, 0, 0, 0 };
        v8i_ gz8 = { 0, 0, 0, 0, 0, 0, 0, 0 };
        __builtin_amdgcn_tensor_load_to_lds(g0, g1, gz4, gz4, gz8, 0);
        __builtin_amdgcn_s_wait_tensorcnt(0);
    }
#else
    for (int i = threadIdx.x; i < KT * 1024; i += blockDim.x)
        reinterpret_cast<unsigned*>(Bs)[i] = reinterpret_cast<const unsigned*>(Bp)[i];
#endif
    __syncthreads();

    int wave = threadIdx.x >> 5;
    int lane = threadIdx.x & 31;
    int tile = blockIdx.x * (blockDim.x >> 5) + wave;
    int tiles = (N + 15) >> 4;
    if (tile < tiles) {                        // wave-uniform: EXEC all-ones inside
        int g = lane >> 4, m = lane & 15;
        int row = tile * 16 + m;
        int rowc = row < N ? row : N - 1;
        const unsigned short* Ar = A + (size_t)rowc * lda + g * 8;
        v8f acc[4] = { v8f{}, v8f{}, v8f{}, v8f{} };
#pragma unroll
        for (int kt = 0; kt < KT; ++kt) {
            union { v16u u; uint4 q[2]; v16bf b; } a;
            a.q[0] = *reinterpret_cast<const uint4*>(Ar + kt * 32);
            a.q[1] = *reinterpret_cast<const uint4*>(Ar + kt * 32 + 16);
#pragma unroll
            for (int j = 0; j < 4; ++j) {
                union { v16u u; v16bf b; } bb;
                bb.u = Bs[(kt * 4 + j) * 32 + lane];
                acc[j] = __builtin_amdgcn_wmma_f32_16x16x32_bf16(
                    false, a.b, false, bb.b, (short)0, acc[j], false, false);
            }
        }
#pragma unroll
        for (int j = 0; j < 4; ++j) {
#pragma unroll
            for (int r = 0; r < 8; ++r) {
                int orow = tile * 16 + r + 8 * g;
                if (orow < N) {
                    int oc = j * 16 + m;
                    float val = acc[j][r] + bias[oc];
                    if (relu) val = fmaxf(val, 0.f);
                    D[(size_t)orow * 64 + oc] = val;
                    if (Dbf) Dbf[(size_t)orow * 64 + oc] = f2bf(val);
                }
            }
        }
    }
}

// m5 gate + gated combine + relu + graphnorm partial sums
__global__ void k_combine(const float* __restrict__ zpart, const float* __restrict__ consts,
                          const float* __restrict__ lin1, const float* __restrict__ lin2,
                          float* __restrict__ hout, float* __restrict__ gn, int N) {
    int n = blockIdx.x * blockDim.x + threadIdx.x;
    if (n >= N) return;
    float gate = sigm(zpart[n] + consts[4]);
    const float* l1 = lin1 + (size_t)n * 64;
    const float* l2 = lin2 + (size_t)n * 64;
    float* ho = hout + (size_t)n * 64;
    float s1 = 0.f, s2 = 0.f;
    for (int j = 0; j < 64; ++j) {
        float v = fmaxf(gate * l1[j] + (1.f - gate) * l2[j], 0.f);
        ho[j] = v; s1 += v; s2 += v * v;
    }
    unsafeAtomicAdd(&gn[0], s1);
    unsafeAtomicAdd(&gn[1], s2);
}

__global__ void k_norm(const float* __restrict__ h, const float* __restrict__ gn,
                       const float* __restrict__ lnw, const float* __restrict__ lnb,
                       float* __restrict__ out, unsigned short* __restrict__ outbf, long M) {
    long i = (long)blockIdx.x * blockDim.x + threadIdx.x;
    if (i >= M) return;
    float inv  = 1.f / (float)M;
    float mean = gn[0] * inv;
    float var  = fmaxf(gn[1] * inv - mean * mean, 0.f);
    float stdv = sqrtf(var);
    int c = (int)(i & 63);
    float v = (h[i] - mean) / (stdv + 1e-5f) * lnw[c] + lnb[c];
    out[i] = v;
    outbf[i] = f2bf(v);
}

// final 64->1 + sigmoid
__global__ void k_head(const float* __restrict__ h, const float* __restrict__ w,
                       const float* __restrict__ b, float* __restrict__ out, int N) {
    int n = blockIdx.x * blockDim.x + threadIdx.x;
    if (n >= N) return;
    const float* hr = h + (size_t)n * 64;
    float z = b[0];
    for (int j = 0; j < 64; ++j) z += w[j] * hr[j];
    out[n] = sigm(z);
}

__global__ void k_final(const float* __restrict__ stats, const float* __restrict__ hh,
                        float* __restrict__ out0) {
    if (threadIdx.x == 0 && blockIdx.x == 0) {
        float Sx = stats[2];
        out0[0] = hypotf(hh[0] * Sx, hh[1] * Sx);
    }
}

static inline void launch_gemm(int KT, int blocks, hipStream_t s,
                               const unsigned short* A, int lda, const unsigned short* Bp,
                               const float* bias, int relu,
                               float* D, unsigned short* Dbf, int N) {
    switch (KT) {
    case 1:  k_gemm<1><<<blocks, 256, 0, s>>>(A, lda, Bp, bias, relu, D, Dbf, N); break;
    case 2:  k_gemm<2><<<blocks, 256, 0, s>>>(A, lda, Bp, bias, relu, D, Dbf, N); break;
    default: k_gemm<7><<<blocks, 256, 0, s>>>(A, lda, Bp, bias, relu, D, Dbf, N); break;
    }
}

// ---------------------------------------------------------------- launcher
extern "C" void kernel_launch(void* const* d_in, const int* in_sizes, int n_in,
                              void* d_out, int out_size, void* d_ws, size_t ws_size,
                              hipStream_t stream) {
    const float* x  = (const float*)d_in[0];
    const int*   ei = (const int*)d_in[1];
    int N = in_sizes[0] / 10;
    int E = in_sizes[1] / 2;
    const int* src = ei;
    const int* dst = ei + E;

    enum { G1B = 0, G1W, G2B, G2W, M2B, M2W, M5B, M5W };
    const float* L[32];
    int cins[3] = { 8, 64, 64 };
    if (n_in >= 34) {
        if (in_sizes[2] == 64) {              // jax sorted-key flatten: g1.b first
            for (int i = 0; i < 24; ++i) L[i] = (const float*)d_in[2 + i];
            L[24] = (const float*)d_in[26];   // ln_b
            L[25] = (const float*)d_in[27];   // ln_w
            for (int i = 0; i < 6; ++i) L[26 + i] = (const float*)d_in[28 + i];
        } else {                              // insertion order: m2.w first (size 28)
            for (int i = 0; i < 3; ++i) {
                int b = 2 + i * 8;
                L[i * 8 + M2W] = (const float*)d_in[b + 0];
                L[i * 8 + M2B] = (const float*)d_in[b + 1];
                L[i * 8 + M5W] = (const float*)d_in[b + 2];
                L[i * 8 + M5B] = (const float*)d_in[b + 3];
                L[i * 8 + G1W] = (const float*)d_in[b + 4];
                L[i * 8 + G1B] = (const float*)d_in[b + 5];
                L[i * 8 + G2W] = (const float*)d_in[b + 6];
                L[i * 8 + G2B] = (const float*)d_in[b + 7];
            }
            L[27] = (const float*)d_in[26]; L[26] = (const float*)d_in[27];
            L[29] = (const float*)d_in[28]; L[28] = (const float*)d_in[29];
            L[31] = (const float*)d_in[30]; L[30] = (const float*)d_in[31];
            L[25] = (const float*)d_in[32]; // ln_w
            L[24] = (const float*)d_in[33]; // ln_b
        }
    } else {
        const float* base = (const float*)d_in[2];
        size_t off = 0;
        for (int i = 0; i < 3; ++i) {
            int C = cins[i], d = 3 * C + 4;
            int sz[8] = { 64, 64 * C, 64, 64 * d, 1, d, 1, d + 2 * C + 1 };
            for (int k = 0; k < 8; ++k) { L[i * 8 + k] = base + off; off += sz[k]; }
        }
        L[24] = base + off; off += 64;
        L[25] = base + off; off += 64;
        int psz[6] = { 64, 4096, 64, 4096, 1, 64 };
        for (int k = 0; k < 6; ++k) { L[26 + k] = base + off; off += psz[k]; }
    }

    // workspace carve-out (256B aligned)
    uintptr_t wp = (uintptr_t)d_ws;
    uintptr_t wend = wp + ws_size;
    auto take = [&](size_t bytes) -> void* {
        void* p = (void*)wp;
        wp = (wp + bytes + 255) & ~(uintptr_t)255;
        return p;
    };
    float* feat   = (float*)take((size_t)N * 64 * 4);
    float* hout   = (float*)take((size_t)N * 64 * 4);
    float* lin1   = (float*)take((size_t)N * 64 * 4);
    float* lin2   = (float*)take((size_t)N * 64 * 4);
    unsigned short* aggr_bf = (unsigned short*)take((size_t)N * 224 * 2);
    unsigned short* xu_bf   = (unsigned short*)take((size_t)N * 64 * 2);
    unsigned short* featbf  = (unsigned short*)take((size_t)N * 64 * 2);
    unsigned short* houtbf  = (unsigned short*)take((size_t)N * 64 * 2);
    float* ssum   = (float*)take((size_t)N * 68 * 4);
    float* cnt    = (float*)take((size_t)N * 4);
    float* dotJ   = (float*)take((size_t)N * 4);
    float* dotI   = (float*)take((size_t)N * 4);
    float* zpart  = (float*)take((size_t)N * 4);
    float* xg     = (float*)take(64 * 4);
    float* stats  = (float*)take(8 * 4);
    float* consts = (float*)take(8 * 4);
    float* gn     = (float*)take(2 * 4);
    unsigned short* packA = (unsigned short*)take(7 * 2048 * 2);
    unsigned short* packB = (unsigned short*)take(7 * 2048 * 2);
    if (wp > wend) return;   // insufficient scratch

    float logc = logf((float)N);
    const int T = 256;
    int tiles = (N + 15) / 16;
    int gemmBlocks = (tiles + 7) / 8;   // 8 waves per block

    k_feat0<<<(N * 8 + T - 1) / T, T, 0, stream>>>(x, feat, N);
    k_stats<<<1, 256, 0, stream>>>(x, N, stats);

    for (int i = 0; i < 3; ++i) {
        int C = cins[i], d = 3 * C + 4;
        int Kp1 = (C + 31) / 32 * 32;
        int Kp2 = (d + 31) / 32 * 32;
        int kt1 = Kp1 / 32, kt2 = Kp2 / 32;
        int tpeShift = (C == 8) ? 3 : 4;
        const float* m2w = L[i * 8 + M2W]; const float* m2b = L[i * 8 + M2B];
        const float* m5w = L[i * 8 + M5W]; const float* m5b = L[i * 8 + M5B];
        const float* g1w = L[i * 8 + G1W]; const float* g1b = L[i * 8 + G1B];
        const float* g2w = L[i * 8 + G2W]; const float* g2b = L[i * 8 + G2B];

        k_colmean<<<C, 256, 0, stream>>>(feat, N, C, xg);
        k_prep<<<1, 1, 0, stream>>>(m2w, m2b, m5w, m5b, xg, stats, C, logc, consts);
        long nss = (long)N * (C + 4);
        k_zero<<<(unsigned)((nss + T - 1) / T), T, 0, stream>>>(ssum, nss);
        k_zero<<<(N + T - 1) / T, T, 0, stream>>>(cnt, N);
        k_zero<<<1, 32, 0, stream>>>(gn, 2);
        k_dotji<<<(N + T - 1) / T, T, 0, stream>>>(feat, m2w, C, N, dotJ, dotI);
        long eth = (long)E << tpeShift;
        k_edge<<<(unsigned)((eth + T - 1) / T), T, 0, stream>>>(
            src, dst, E, x, feat, C, dotJ, dotI, consts, ssum, cnt, tpeShift);
        k_aggr<<<(N + T - 1) / T, T, 0, stream>>>(
            ssum, cnt, feat, xg, x, m5w, C, N, Kp2, Kp1, logc, aggr_bf, xu_bf, zpart);
        k_pack<<<(kt1 * 2048 + T - 1) / T, T, 0, stream>>>(g1w, C, kt1, packA);
        k_pack<<<(kt2 * 2048 + T - 1) / T, T, 0, stream>>>(g2w, d, kt2, packB);
        launch_gemm(kt1, gemmBlocks, stream, xu_bf,   Kp1, packA, g1b, 0, lin1, nullptr, N);
        launch_gemm(kt2, gemmBlocks, stream, aggr_bf, Kp2, packB, g2b, 0, lin2, nullptr, N);
        k_combine<<<(N + T - 1) / T, T, 0, stream>>>(zpart, consts, lin1, lin2, hout, gn, N);
        long M = (long)N * 64;
        k_norm<<<(unsigned)((M + T - 1) / T), T, 0, stream>>>(
            hout, gn, L[25], L[24], feat, featbf, M);
    }

    // post MLP: 64->64 relu, 64->64 relu, 64->1 sigmoid
    k_pack<<<(2 * 2048 + T - 1) / T, T, 0, stream>>>(L[27], 64, 2, packA);
    launch_gemm(2, gemmBlocks, stream, featbf, 64, packA, L[26], 1, hout, houtbf, N);
    k_pack<<<(2 * 2048 + T - 1) / T, T, 0, stream>>>(L[29], 64, 2, packA);
    launch_gemm(2, gemmBlocks, stream, houtbf, 64, packA, L[28], 1, lin1, nullptr, N);
    k_head<<<(N + T - 1) / T, T, 0, stream>>>(lin1, L[31], L[30], (float*)d_out + 1, N);
    k_final<<<1, 32, 0, stream>>>(stats, (float*)d_out + 1, (float*)d_out);
}
